// BidirectionalAttention_6382321402532
// MI455X (gfx1250) — matrix operand order repaired and checked by reading, not compiled, not run on hardware
//
#include <hip/hip_runtime.h>
#include <hip/hip_bf16.h>

typedef __attribute__((ext_vector_type(16))) _Float16 v16h;
typedef __attribute__((ext_vector_type(8)))  _Float16 v8h;
typedef __attribute__((ext_vector_type(4)))  _Float16 v4h;
typedef __attribute__((ext_vector_type(8)))  float    v8f;
typedef __attribute__((ext_vector_type(4)))  unsigned u32x4;
typedef __attribute__((ext_vector_type(8)))  unsigned u32x8;

#define Bq   4
#define Tq   2048
#define Cq   2048
#define Hq   16
#define Dq   128
#define Mq   (Bq * Tq)   // 8192

// ---------------------------------------------------------------------------
// LDS byte address of a __shared__-derived pointer: the LDS flat aperture
// keeps the wave-relative LDS offset in addr[31:0] (ISA 10.2 aperture calc).
__device__ __forceinline__ unsigned lds_off(const void* p) {
    return (unsigned)(unsigned long long)p;
}

// Async DMA: global -> LDS, 16B per lane, tracked by ASYNCcnt.
__device__ __forceinline__ void async_g2l_b128(unsigned lds, const void* gptr) {
    asm volatile("global_load_async_to_lds_b128 %0, %1, off"
                 :: "v"(lds), "v"(gptr) : "memory");
}
__device__ __forceinline__ void wait_async0() {
    asm volatile("s_wait_asynccnt 0x0" ::: "memory");
}

// Fragment loaders (A: lane=row, split K pattern; B: lane=col, split K pattern)
__device__ __forceinline__ v16h load_frag_a(const _Float16* row, int hi) {
    union { v16h v; v8h h[2]; } u;
    u.h[0] = *(const v8h*)(row + hi * 8);
    u.h[1] = *(const v8h*)(row + hi * 8 + 16);
    return u.v;
}
__device__ __forceinline__ v16h load_frag_b(const _Float16* col, int hi) {
    union { v16h v; v8h h[2]; } u;
    u.h[0] = *(const v8h*)(col + hi * 8);
    u.h[1] = *(const v8h*)(col + hi * 8 + 16);
    return u.v;
}
__device__ __forceinline__ v8f wmma_f16(v16h a, v16h b, v8f c) {
    return __builtin_amdgcn_wmma_f32_16x16x32_f16(false, a, false, b,
                                                  (short)0, c, false, false);
}

// ---------------------------------------------------------------------------
__global__ __launch_bounds__(256) void cvt_f32_f16(const float* __restrict__ src,
                                                   _Float16* __restrict__ dst,
                                                   int n4) {
    int i = blockIdx.x * 256 + threadIdx.x;
    if (i < n4) {
        float4 v = ((const float4*)src)[i];
        v4h o;
        o.x = (_Float16)v.x; o.y = (_Float16)v.y;
        o.z = (_Float16)v.z; o.w = (_Float16)v.w;
        ((v4h*)dst)[i] = o;
    }
}

// ---------------------------------------------------------------------------
// GEMM: Y[m,n] = sum_k A[m,k] * Bm[n,k]  (A @ Bm^T), f16 in, f32 accum.
// WG tile 128x128, K-step 32, double-buffered async global->LDS staging.
// 8 waves in 2(M) x 4(N); each wave: 4(M) x 2(N) = 8 WMMA frags.
__global__ __launch_bounds__(256) void gemm_xt(const _Float16* __restrict__ A,
                                               const _Float16* __restrict__ Bm,
                                               void* __restrict__ Cout,
                                               int Mdim, int Ndim, int Kdim,
                                               int store_f32) {
    __shared__ __align__(16) _Float16 As[2][128][32];
    __shared__ __align__(16) _Float16 Bs[2][128][32];

    const int tid  = threadIdx.x;
    const int wave = tid >> 5;
    const int lane = tid & 31;
    const int l16  = lane & 15;
    const int hi   = lane >> 4;
    const int m0   = blockIdx.x * 128;
    const int n0   = blockIdx.y * 128;
    const int wm   = wave >> 2;      // 0..1 -> 64-row half
    const int wn   = wave & 3;       // 0..3 -> 32-col chunk

    const unsigned asBase = lds_off(&As[0][0][0]);
    const unsigned bsBase = lds_off(&Bs[0][0][0]);

    const v8f zero = {0.f,0.f,0.f,0.f,0.f,0.f,0.f,0.f};
    v8f acc[4][2] = {{zero, zero}, {zero, zero}, {zero, zero}, {zero, zero}};

    // stage one 128x32 K-slab of A and B: 2 x 16B chunks per thread each
    auto issue_tile = [&](int k0, int buf) {
#pragma unroll
        for (int j = 0; j < 2; ++j) {
            int chunk = tid * 2 + j;              // 0..511
            int row = chunk >> 2, seg = chunk & 3;
            unsigned lo = (unsigned)(buf * 128 * 32 + row * 32 + seg * 8) * 2u;
            async_g2l_b128(asBase + lo,
                           A  + (size_t)(m0 + row) * Kdim + k0 + seg * 8);
            async_g2l_b128(bsBase + lo,
                           Bm + (size_t)(n0 + row) * Kdim + k0 + seg * 8);
        }
    };

    const int nIter = Kdim / 32;
    issue_tile(0, 0);
    for (int it = 0; it < nIter; ++it) {
        wait_async0();
        __syncthreads();
        int buf = it & 1;
        if (it + 1 < nIter) issue_tile((it + 1) * 32, buf ^ 1);

        v16h bf[2];
#pragma unroll
        for (int g = 0; g < 2; ++g)
            bf[g] = load_frag_b(&Bs[buf][wn * 32 + g * 16 + l16][0], hi);
#pragma unroll
        for (int f = 0; f < 4; ++f) {
            v16h af = load_frag_a(&As[buf][wm * 64 + f * 16 + l16][0], hi);
#pragma unroll
            for (int g = 0; g < 2; ++g)
                acc[f][g] = wmma_f16(af, bf[g], acc[f][g]);
        }
    }

#pragma unroll
    for (int f = 0; f < 4; ++f)
#pragma unroll
        for (int g = 0; g < 2; ++g)
#pragma unroll
            for (int r = 0; r < 8; ++r) {
                int row = m0 + wm * 64 + f * 16 + r + hi * 8;
                int col = n0 + wn * 32 + g * 16 + l16;
                float v = acc[f][g][r];
                if (store_f32)
                    ((float*)Cout)[(size_t)row * Ndim + col] = v;
                else
                    ((_Float16*)Cout)[(size_t)row * Ndim + col] = (_Float16)v;
            }
}

// ---------------------------------------------------------------------------
__global__ __launch_bounds__(256) void rotary_f16(_Float16* __restrict__ q,
                                                  const float* __restrict__ cs,
                                                  const float* __restrict__ sn) {
    size_t i = (size_t)blockIdx.x * 256 + threadIdx.x;  // over B*T*H*64
    int d = (int)(i & 63);
    size_t rem = i >> 6;
    int h = (int)(rem & (Hq - 1));
    size_t bt = rem >> 4;                 // b*T + t
    int t = (int)(bt & (Tq - 1));
    size_t base = bt * (size_t)Cq + (size_t)h * Dq;
    float x1 = (float)q[base + d];
    float x2 = (float)q[base + 64 + d];
    float c = cs[t * 64 + d];
    float s = sn[t * 64 + d];
    q[base + d]      = (_Float16)(x1 * c + x2 * s);
    q[base + 64 + d] = (_Float16)(x1 * c - x2 * s);
}

// ---------------------------------------------------------------------------
// Flash attention. Grid: (b*h, T/128). 8 waves; wave owns 16 q-rows with a
// 16x128 f32 accumulator. 32 keys per step. K-tile fetched by the Tensor
// Data Mover (one descriptor-driven DMA), V-tile transposed manually.
__global__ __launch_bounds__(256) void flash_attn(const _Float16* __restrict__ qh,
                                                  const _Float16* __restrict__ kh,
                                                  const _Float16* __restrict__ vh,
                                                  _Float16* __restrict__ yh) {
    __shared__ __align__(16) _Float16 Ks[32][128];   // keys x d (TDM dest)
    __shared__ __align__(16) _Float16 Vt[128][32];   // d x keys
    __shared__ __align__(16) _Float16 Pb[8][16][32]; // per-wave P bounce

    const int bh   = blockIdx.x;
    const int b    = bh >> 4;
    const int h    = bh & 15;
    const int qt   = blockIdx.y;
    const int tid  = threadIdx.x;
    const int wave = tid >> 5;
    const int lane = tid & 31;
    const int l16  = lane & 15;
    const int hi   = lane >> 4;
    const float scale = 0.08838834764831845f;        // 1/sqrt(128)
    const unsigned ksBase = lds_off(&Ks[0][0]);

    const int qrow = qt * 128 + wave * 16 + l16;
    const _Float16* qp = qh + ((size_t)(b * Tq + qrow)) * Cq + h * Dq;
    v16h qf[4];
#pragma unroll
    for (int c = 0; c < 4; ++c) qf[c] = load_frag_a(qp + c * 32, hi);

    const v8f zero = {0.f,0.f,0.f,0.f,0.f,0.f,0.f,0.f};
    v8f acc[8] = {zero, zero, zero, zero, zero, zero, zero, zero};
    float mrow[8], lrow[8];
#pragma unroll
    for (int r = 0; r < 8; ++r) { mrow[r] = -1e30f; lrow[r] = 0.f; }

    for (int j0 = 0; j0 < Tq; j0 += 32) {
        __syncthreads();                 // previous tile fully consumed
        if (wave == 0) {
            // TDM descriptor (ISA 8.3/8.4): 2-D tile 128(d) x 32(keys),
            // element 2B, row stride 2048 elements.
            unsigned long long ga = (unsigned long long)
                (kh + ((size_t)(b * Tq + j0)) * Cq + h * Dq);
            u32x4 g0;
            g0[0] = 1u;                                   // count=1, user mode
            g0[1] = ksBase;                               // lds_addr
            g0[2] = (unsigned)ga;                         // global_addr[31:0]
            g0[3] = (unsigned)((ga >> 32) & 0x01FFFFFFu)  // global_addr[56:32]
                    | (2u << 30);                         // type=2 (image)
            u32x8 g1;
            g1[0] = 1u << 16;            // data_size=1 (2 bytes)
            g1[1] = 128u << 16;          // tensor_dim0 = 128 (bits 79:48 lo)
            g1[2] = 32u << 16;           // tensor_dim1 = 32  (bits 111:80 lo)
            g1[3] = 128u << 16;          // tile_dim0 = 128   (bits 127:112)
            g1[4] = 32u;                 // tile_dim1 = 32    (bits 143:128)
            g1[5] = 2048u;               // tensor_dim0_stride = 2048
            g1[6] = 0u;
            g1[7] = 0u;
            asm volatile("tensor_load_to_lds %0, %1"
                         :: "s"(g0), "s"(g1) : "memory");
        }
        // stage V transposed (d x key), 16 f16 per thread
#pragma unroll
        for (int gg = 0; gg < 2; ++gg) {
            int g = tid * 2 + gg;                    // 0..511
            int key = g >> 4, dseg = g & 15;
            v8h val = *(const v8h*)(vh + ((size_t)(b * Tq + j0 + key)) * Cq +
                                    h * Dq + dseg * 8);
#pragma unroll
            for (int e = 0; e < 8; ++e) Vt[dseg * 8 + e][key] = val[e];
        }
        if (wave == 0) __builtin_amdgcn_s_wait_tensorcnt(0);
        __syncthreads();

        // S = Q * K^T for 32 keys (two 16x16 tiles)
        v8f s0 = zero, s1 = zero;
#pragma unroll
        for (int c = 0; c < 4; ++c) {
            v16h k0f = load_frag_b(&Ks[l16][c * 32], hi);
            v16h k1f = load_frag_b(&Ks[16 + l16][c * 32], hi);
            s0 = wmma_f16(qf[c], k0f, s0);
            s1 = wmma_f16(qf[c], k1f, s1);
        }

        // online softmax (row stats live in 16-lane halves)
#pragma unroll
        for (int r = 0; r < 8; ++r) {
            float a0 = s0[r] * scale;
            float a1 = s1[r] * scale;
            float cm = fmaxf(a0, a1);
            cm = fmaxf(cm, __shfl_xor(cm, 1));
            cm = fmaxf(cm, __shfl_xor(cm, 2));
            cm = fmaxf(cm, __shfl_xor(cm, 4));
            cm = fmaxf(cm, __shfl_xor(cm, 8));
            float mnew  = fmaxf(mrow[r], cm);
            float alpha = __expf(mrow[r] - mnew);
            float p0 = __expf(a0 - mnew);
            float p1 = __expf(a1 - mnew);
            float rs = p0 + p1;
            rs += __shfl_xor(rs, 1);
            rs += __shfl_xor(rs, 2);
            rs += __shfl_xor(rs, 4);
            rs += __shfl_xor(rs, 8);
            lrow[r] = lrow[r] * alpha + rs;
            mrow[r] = mnew;
#pragma unroll
            for (int f = 0; f < 8; ++f) acc[f][r] *= alpha;
            int prow = r + hi * 8;
            Pb[wave][prow][l16]      = (_Float16)p0;
            Pb[wave][prow][16 + l16] = (_Float16)p1;
        }

        // O += P (16x32) * V (32x128)
        v16h pf = load_frag_a(&Pb[wave][l16][0], hi);
#pragma unroll
        for (int f = 0; f < 8; ++f) {
            v16h vf = load_frag_b(&Vt[f * 16 + l16][0], hi);
            acc[f] = wmma_f16(pf, vf, acc[f]);
        }
    }

#pragma unroll
    for (int f = 0; f < 8; ++f)
#pragma unroll
        for (int r = 0; r < 8; ++r) {
            int row = qt * 128 + wave * 16 + r + hi * 8;
            float v = acc[f][r] / lrow[r];
            yh[((size_t)(b * Tq + row)) * Cq + h * Dq + f * 16 + l16] =
                (_Float16)v;
        }
}

// ---------------------------------------------------------------------------
extern "C" void kernel_launch(void* const* d_in, const int* in_sizes, int n_in,
                              void* d_out, int out_size, void* d_ws, size_t ws_size,
                              hipStream_t stream) {
    (void)in_sizes; (void)n_in; (void)out_size; (void)ws_size;
    const float* x    = (const float*)d_in[0];
    const float* cosp = (const float*)d_in[1];
    const float* sinp = (const float*)d_in[2];
    const float* wq   = (const float*)d_in[3];
    const float* wk   = (const float*)d_in[4];
    const float* wv   = (const float*)d_in[5];
    const float* wo   = (const float*)d_in[6];
    float* out = (float*)d_out;

    _Float16* xh  = (_Float16*)d_ws;
    _Float16* wqh = xh  + (size_t)Mq * Cq;
    _Float16* wkh = wqh + (size_t)Cq * Cq;
    _Float16* wvh = wkh + (size_t)Cq * Cq;
    _Float16* woh = wvh + (size_t)Cq * Cq;
    _Float16* qh  = woh + (size_t)Cq * Cq;
    _Float16* kh  = qh  + (size_t)Mq * Cq;
    _Float16* vh  = kh  + (size_t)Mq * Cq;
    _Float16* yh  = vh  + (size_t)Mq * Cq;

    const int nX4 = (Mq * Cq) / 4;
    const int nW4 = (Cq * Cq) / 4;
    cvt_f32_f16<<<nX4 / 256, 256, 0, stream>>>(x,  xh,  nX4);
    cvt_f32_f16<<<nW4 / 256, 256, 0, stream>>>(wq, wqh, nW4);
    cvt_f32_f16<<<nW4 / 256, 256, 0, stream>>>(wk, wkh, nW4);
    cvt_f32_f16<<<nW4 / 256, 256, 0, stream>>>(wv, wvh, nW4);
    cvt_f32_f16<<<nW4 / 256, 256, 0, stream>>>(wo, woh, nW4);

    dim3 ggrid(Mq / 128, Cq / 128);                 // 64 x 16
    gemm_xt<<<ggrid, 256, 0, stream>>>(xh, wqh, qh, Mq, Cq, Cq, 0);
    gemm_xt<<<ggrid, 256, 0, stream>>>(xh, wkh, kh, Mq, Cq, Cq, 0);
    gemm_xt<<<ggrid, 256, 0, stream>>>(xh, wvh, vh, Mq, Cq, Cq, 0);

    const int nPairs = Bq * Tq * Hq * 64;
    rotary_f16<<<nPairs / 256, 256, 0, stream>>>(qh, cosp, sinp);
    rotary_f16<<<nPairs / 256, 256, 0, stream>>>(kh, cosp, sinp);

    dim3 agrid(Bq * Hq, Tq / 128);                  // 64 x 16
    flash_attn<<<agrid, 256, 0, stream>>>(qh, kh, vh, yh);

    gemm_xt<<<ggrid, 256, 0, stream>>>(yh, woh, out, Mq, Cq, Cq, 1);
}
